// CAM_40029095199434
// MI455X (gfx1250) — matrix operand order repaired and checked by reading, not compiled, not run on hardware
//
#include <hip/hip_runtime.h>

// CAM (channel attention) for MI455X / gfx1250, wave32 + WMMA bf16.
// B=16, C=512, N=H*W=4096.
//  c1: xl -> bf16 row-major        (A of GEMM1)
//  c2: xh -> bf16 row-major + bf16 transposed [N][C]  (B of GEMM1 / GEMM2)
//  k1: energy = Q*K^T   NT GEMM, 64x64 wave tiles, pure load+wmma loop
//  k2: attn = softmax(-energy) row-wise, stored bf16 (A of GEMM2)
//  k3: out  = gamma*(attn*K) + x_low   NN GEMM via xh_t, 64x64 wave tiles

#define BSZ  16
#define CDIM 512
#define NDIM 4096

typedef __attribute__((ext_vector_type(16))) __bf16 v16bf;
typedef __attribute__((ext_vector_type(8)))  float  v8f;

union FragAB {
  v16bf v;
  unsigned u[8];
};

__device__ __forceinline__ unsigned bf16pack(float lo, float hi) {
  // round-to-nearest-even fp32 -> bf16, packed pair (lo in [15:0])
  unsigned a = __float_as_uint(lo);
  unsigned b = __float_as_uint(hi);
  a = (a + 0x7FFFu + ((a >> 16) & 1u)) >> 16;
  b = (b + 0x7FFFu + ((b >> 16) & 1u)) >> 16;
  return a | (b << 16);
}

__device__ __forceinline__ unsigned short f2bf(float x) {
  unsigned u = __float_as_uint(x);
  return (unsigned short)((u + 0x7FFFu + ((u >> 16) & 1u)) >> 16);
}

// ---------------------------------------------------------------------------
// c1: straight fp32 -> bf16 stream (8 elements / thread)
// ---------------------------------------------------------------------------
__global__ __launch_bounds__(256) void cam_cvt_stream(
    const float* __restrict__ src, unsigned short* __restrict__ dst) {
  const size_t base = ((size_t)blockIdx.x * 256 + threadIdx.x) * 8;
  const float4 f0 = *(const float4*)(src + base);
  const float4 f1 = *(const float4*)(src + base + 4);
  uint4 o;
  o.x = bf16pack(f0.x, f0.y);
  o.y = bf16pack(f0.z, f0.w);
  o.z = bf16pack(f1.x, f1.y);
  o.w = bf16pack(f1.z, f1.w);
  *(uint4*)(dst + base) = o;
}

// ---------------------------------------------------------------------------
// c2: xh fp32 [B][C][N] -> xh_bf bf16 [B][C][N]  +  xh_t bf16 [B][N][C]
// 64x64 tile per block; transpose through LDS (rows padded to 68 ushorts).
// ---------------------------------------------------------------------------
__global__ __launch_bounds__(256) void cam_cvt_xh(
    const float* __restrict__ xh, unsigned short* __restrict__ xh_bf,
    unsigned short* __restrict__ xh_t) {
  __shared__ __align__(16) unsigned short lds[64][68];  // [n][j]
  const int b  = blockIdx.y;
  const int jb = (blockIdx.x & 7) * 64;   // channel tile
  const int nb = (blockIdx.x >> 3) * 64;  // pixel tile

#pragma unroll
  for (int i = 0; i < 4; ++i) {
    const int q = threadIdx.x + i * 256;
    const int j = q >> 4;          // 0..63
    const int n4 = (q & 15) * 4;   // 0..60
    const size_t off = ((size_t)b * CDIM + jb + j) * NDIM + nb + n4;
    const float4 f = *(const float4*)(xh + off);
    uint2 o;
    o.x = bf16pack(f.x, f.y);
    o.y = bf16pack(f.z, f.w);
    *(uint2*)(xh_bf + off) = o;
    lds[n4 + 0][j] = f2bf(f.x);
    lds[n4 + 1][j] = f2bf(f.y);
    lds[n4 + 2][j] = f2bf(f.z);
    lds[n4 + 3][j] = f2bf(f.w);
  }
  __syncthreads();
#pragma unroll
  for (int i = 0; i < 4; ++i) {
    const int q = threadIdx.x + i * 256;
    const int n = q >> 4;
    const int j4 = (q & 15) * 4;
    const uint2 v = *(const uint2*)&lds[n][j4];
    *(uint2*)(xh_t + ((size_t)b * NDIM + nb + n) * CDIM + jb + j4) = v;
  }
}

// ---------------------------------------------------------------------------
// k1: energy[b,i,j] = sum_n xl[b,i,n]*xh[b,j,n]  (NT GEMM, all-bf16 inputs)
// Block = 256 (8 waves); each wave an independent 64x64 tile (4x4 frags).
// A-frag lane: row tm+mi*16+l16, 8-elem chunks at kb+h*8 and kb+16+h*8.
// B-frag lane: K row tn+ni*16+l16, 8-elem chunks at kb+h*8 and kb+16+h*8.
// ---------------------------------------------------------------------------
__global__ __launch_bounds__(256) void cam_energy_gemm(
    const unsigned short* __restrict__ qbf, const unsigned short* __restrict__ kbf,
    float* __restrict__ energy) {
  const int b    = blockIdx.y;
  const int wave = threadIdx.x >> 5;
  const int lane = threadIdx.x & 31;
  const int half = lane >> 4;
  const int l16  = lane & 15;
  const int t    = blockIdx.x * 8 + wave;  // 0..63
  const int tm   = (t >> 3) * 64;
  const int tn   = (t & 7) * 64;

  const unsigned short* Q = qbf + (size_t)b * CDIM * NDIM;
  const unsigned short* K = kbf + (size_t)b * CDIM * NDIM;

  const unsigned short* qrow[4];
  const unsigned short* krow[4];
#pragma unroll
  for (int i = 0; i < 4; ++i) {
    qrow[i] = Q + (size_t)(tm + i * 16 + l16) * NDIM;
    krow[i] = K + (size_t)(tn + i * 16 + l16) * NDIM;
  }

  v8f acc[4][4] = {};

  for (int kb = 0; kb < NDIM; kb += 32) {
    FragAB a[4], bb[4];
#pragma unroll
    for (int i = 0; i < 4; ++i) {
      const unsigned short* pa = qrow[i] + kb + half * 8;
      *(uint4*)&a[i].u[0] = *(const uint4*)pa;
      *(uint4*)&a[i].u[4] = *(const uint4*)(pa + 16);
      const unsigned short* pb = krow[i] + kb + half * 8;
      *(uint4*)&bb[i].u[0] = *(const uint4*)pb;
      *(uint4*)&bb[i].u[4] = *(const uint4*)(pb + 16);
    }
#pragma unroll
    for (int mi = 0; mi < 4; ++mi)
#pragma unroll
      for (int ni = 0; ni < 4; ++ni)
        acc[mi][ni] = __builtin_amdgcn_wmma_f32_16x16x32_bf16(
            false, a[mi].v, false, bb[ni].v, (short)0, acc[mi][ni], false,
            false);
  }

  // C/D layout: VGPR r holds M = half*8 + r, N = l16
#pragma unroll
  for (int mi = 0; mi < 4; ++mi)
#pragma unroll
    for (int r = 0; r < 8; ++r) {
      const int row = tm + mi * 16 + half * 8 + r;
#pragma unroll
      for (int ni = 0; ni < 4; ++ni)
        energy[((size_t)b * CDIM + row) * CDIM + tn + ni * 16 + l16] =
            acc[mi][ni][r];
    }
}

// ---------------------------------------------------------------------------
// k2: attn = softmax(max-e) = softmax(-e); stable: exp(min-e)/sum. bf16 out.
// One wave per 512-element row.
// ---------------------------------------------------------------------------
__global__ __launch_bounds__(256) void cam_softmax(
    const float* __restrict__ energy, unsigned short* __restrict__ attn) {
  const int row  = blockIdx.x * 8 + (threadIdx.x >> 5);
  const int lane = threadIdx.x & 31;
  const float* e = energy + (size_t)row * CDIM;

  float v[16];
  float mn = 3.4e38f;
#pragma unroll
  for (int t = 0; t < 16; ++t) {
    v[t] = e[lane + t * 32];
    mn   = fminf(mn, v[t]);
  }
#pragma unroll
  for (int s = 16; s; s >>= 1) mn = fminf(mn, __shfl_xor(mn, s, 32));

  float sum = 0.f;
#pragma unroll
  for (int t = 0; t < 16; ++t) {
    v[t] = __expf(mn - v[t]);
    sum += v[t];
  }
#pragma unroll
  for (int s = 16; s; s >>= 1) sum += __shfl_xor(sum, s, 32);
  const float inv = 1.0f / sum;

  unsigned short* o = attn + (size_t)row * CDIM;
#pragma unroll
  for (int t = 0; t < 16; ++t) o[lane + t * 32] = f2bf(v[t] * inv);
}

// ---------------------------------------------------------------------------
// k3: out[b,i,n] = gamma*sum_j attn[b,i,j]*xh[b,j,n] + xl[b,i,n]
// NN GEMM using xh_t [N][C]: B-frag lane = xh_t row nt+ni*16+l16,
// 8-elem chunks at kb+h*8 and kb+16+h*8 -> no LDS, pure load+wmma loop.
// Block = 256 (8 waves); each wave a 64(M)x64(N) tile (4x4 frags).
// ---------------------------------------------------------------------------
__global__ __launch_bounds__(256) void cam_out_gemm(
    const unsigned short* __restrict__ attn,
    const unsigned short* __restrict__ xh_t, const float* __restrict__ xl,
    const float* __restrict__ gamma, float* __restrict__ out) {
  const int b    = blockIdx.y;
  const int wave = threadIdx.x >> 5;
  const int lane = threadIdx.x & 31;
  const int half = lane >> 4;
  const int l16  = lane & 15;
  const int t    = blockIdx.x * 8 + wave;  // 0..511
  const int mt   = (t & 7) * 64;           // channel tile (M)
  const int nt   = (t >> 3) * 64;          // pixel tile (N)

  const unsigned short* arow[4];
  const unsigned short* brow[4];
#pragma unroll
  for (int i = 0; i < 4; ++i) {
    arow[i] = attn + ((size_t)b * CDIM + mt + i * 16 + l16) * CDIM;
    brow[i] = xh_t + ((size_t)b * NDIM + nt + i * 16 + l16) * CDIM;
  }

  v8f acc[4][4] = {};

  for (int kb = 0; kb < CDIM; kb += 32) {
    FragAB a[4], bb[4];
#pragma unroll
    for (int i = 0; i < 4; ++i) {
      const unsigned short* pa = arow[i] + kb + half * 8;
      *(uint4*)&a[i].u[0] = *(const uint4*)pa;
      *(uint4*)&a[i].u[4] = *(const uint4*)(pa + 16);
      const unsigned short* pb = brow[i] + kb + half * 8;
      *(uint4*)&bb[i].u[0] = *(const uint4*)pb;
      *(uint4*)&bb[i].u[4] = *(const uint4*)(pb + 16);
    }
#pragma unroll
    for (int mi = 0; mi < 4; ++mi)
#pragma unroll
      for (int ni = 0; ni < 4; ++ni)
        acc[mi][ni] = __builtin_amdgcn_wmma_f32_16x16x32_bf16(
            false, a[mi].v, false, bb[ni].v, (short)0, acc[mi][ni], false,
            false);
  }

  const float g = gamma[0];
#pragma unroll
  for (int mi = 0; mi < 4; ++mi)
#pragma unroll
    for (int r = 0; r < 8; ++r) {
      const int row = mt + mi * 16 + half * 8 + r;
#pragma unroll
      for (int ni = 0; ni < 4; ++ni) {
        const int    col = nt + ni * 16 + l16;
        const size_t idx = ((size_t)b * CDIM + row) * NDIM + col;
        out[idx] = g * acc[mi][ni][r] + xl[idx];
      }
    }
}

extern "C" void kernel_launch(void* const* d_in, const int* in_sizes, int n_in,
                              void* d_out, int out_size, void* d_ws,
                              size_t ws_size, hipStream_t stream) {
  (void)in_sizes; (void)n_in; (void)out_size; (void)ws_size;
  const float* xh    = (const float*)d_in[0];  // x_high
  const float* xl    = (const float*)d_in[1];  // x_low
  const float* gamma = (const float*)d_in[2];
  float* out = (float*)d_out;

  const size_t NELEM = (size_t)BSZ * CDIM * NDIM;  // 33.5M
  char* ws = (char*)d_ws;
  unsigned short* xl_bf  = (unsigned short*)(ws);                  // 64 MiB
  unsigned short* xh_bf  = (unsigned short*)(ws + NELEM * 2);      // 64 MiB
  unsigned short* xh_t   = (unsigned short*)(ws + NELEM * 4);      // 64 MiB
  float*          energy = (float*)(ws + NELEM * 6);               // 16 MiB
  unsigned short* attn   = (unsigned short*)(ws + NELEM * 6 +
                               (size_t)BSZ * CDIM * CDIM * 4);     // 8 MiB

  // c1: x_low -> bf16
  cam_cvt_stream<<<dim3(NELEM / 2048), 256, 0, stream>>>(xl, xl_bf);
  // c2: x_high -> bf16 row-major + transposed
  cam_cvt_xh<<<dim3((CDIM / 64) * (NDIM / 64), BSZ), 256, 0, stream>>>(
      xh, xh_bf, xh_t);
  // k1: energy GEMM (8 blocks x 16 batches, 64x64 tile per wave)
  cam_energy_gemm<<<dim3((CDIM / 64) * (CDIM / 64) / 8, BSZ), 256, 0,
                    stream>>>(xl_bf, xh_bf, energy);
  // k2: softmax
  cam_softmax<<<dim3((BSZ * CDIM) / 8), 256, 0, stream>>>(energy, attn);
  // k3: out GEMM + epilogue (64 blocks x 16 batches)
  cam_out_gemm<<<dim3((CDIM / 64) * (NDIM / 64) / 8, BSZ), 256, 0, stream>>>(
      attn, xh_t, xl, gamma, out);
}
